// RnnModel2_63582695850443
// MI455X (gfx1250) — matrix operand order repaired and checked by reading, not compiled, not run on hardware
//
#include <hip/hip_runtime.h>
#include <math.h>

// Problem constants (match reference)
#define EE 512   // embedding
#define HH 1024  // hidden
#define NCLS 20  // classes
#define BB 64    // batch
#define TT 512   // seq len

typedef __attribute__((ext_vector_type(16))) __bf16 v16bf;
typedef __attribute__((ext_vector_type(8)))  __bf16 v8bf;
typedef __attribute__((ext_vector_type(8)))  float  v8f;

__device__ __forceinline__ unsigned short f32_to_bf16_raw(float f) {
  unsigned int u = __float_as_uint(f);
  u += 0x7FFFu + ((u >> 16) & 1u);   // round-to-nearest-even
  return (unsigned short)(u >> 16);
}

// ---- WMMA operand loaders (wave32, contiguous K order k = 16h + s, same in A and B) ----
// A: 16x32 bf16, row-major source with leading dim ld (elements).
// lane l<16 holds M=l, K=0..15; lanes 16..31 hold M=l-16, K=16..31
__device__ __forceinline__ v16bf load_a_tile(const unsigned short* __restrict__ base,
                                             int ld, int k0) {
  const int l  = threadIdx.x & 31;
  const int m  = l & 15;
  const int ko = (l >> 4) * 16;
  const unsigned short* p = base + (size_t)m * ld + k0;
  v16bf a;
  *(v8bf*)&a       = *(const v8bf*)(p + ko);
  *((v8bf*)&a + 1) = *(const v8bf*)(p + 8 + ko);
  return a;
}

// B: 32x16 bf16 (KxN). Weight stored row-major [N][K] => B(k,n)=W[n0+n][k],
// contiguous in K per lane: lanes 0-15 K=0..15 (n=lane), lanes 16-31 K=16..31.
__device__ __forceinline__ v16bf load_b_tile(const unsigned short* __restrict__ wrow0,
                                             int ld, int k0) {
  const int l  = threadIdx.x & 31;
  const int n  = l & 15;
  const int kb = (l >> 4) * 16;
  return *(const v16bf*)(wrow0 + (size_t)n * ld + k0 + kb);
}

// ---------------------------------------------------------------------------
// Kernel 1: per-launch init — zero barrier counters & h0, convert weights and
// transpose x [B][T][E] -> bf16 [T][B][E].
// ---------------------------------------------------------------------------
__global__ void init_convert(const float* __restrict__ x_in,
                             const float* __restrict__ wih,
                             const float* __restrict__ whh,
                             const float* __restrict__ fc1w,
                             unsigned short* __restrict__ xb,
                             unsigned short* __restrict__ wihb,
                             unsigned short* __restrict__ whhb,
                             unsigned short* __restrict__ fc1wb,
                             unsigned short* __restrict__ h0buf,
                             int* __restrict__ counters) {
  const size_t tid    = (size_t)blockIdx.x * blockDim.x + threadIdx.x;
  const size_t stride = (size_t)gridDim.x * blockDim.x;
  for (size_t i = tid; i < (size_t)TT; i += stride)      counters[i] = 0;
  for (size_t i = tid; i < (size_t)BB * HH; i += stride) h0buf[i] = 0;
  for (size_t i = tid; i < (size_t)HH * EE; i += stride) wihb[i]  = f32_to_bf16_raw(wih[i]);
  for (size_t i = tid; i < (size_t)HH * HH; i += stride) whhb[i]  = f32_to_bf16_raw(whh[i]);
  for (size_t i = tid; i < (size_t)HH * HH; i += stride) fc1wb[i] = f32_to_bf16_raw(fc1w[i]);
  for (size_t i = tid; i < (size_t)BB * TT * EE; i += stride) {
    size_t e  = i % EE;
    size_t tb = i / EE;
    size_t b  = tb % BB;
    size_t t  = tb / BB;
    xb[i] = f32_to_bf16_raw(x_in[(b * TT + t) * EE + e]);
  }
}

// ---------------------------------------------------------------------------
// Kernel 2: x_proj[TB][H] = x_bf16[TB][E] @ W_ih^T + (b_ih + b_hh)   (fp32 out)
// block = 128 threads (4 waves), 64(M)x64(N) tile per block.
// ---------------------------------------------------------------------------
__global__ void __launch_bounds__(128)
xproj_gemm(const unsigned short* __restrict__ xb,
           const unsigned short* __restrict__ wih,
           const float* __restrict__ b_ih,
           const float* __restrict__ b_hh,
           float* __restrict__ xproj) {
  const int w  = threadIdx.x >> 5;
  const int l  = threadIdx.x & 31;
  const int m0 = blockIdx.x * 64 + w * 16;
  const int n0 = blockIdx.y * 64;
  const int nlane = l & 15;

  v8f acc[4];
  #pragma unroll
  for (int nt = 0; nt < 4; ++nt) {
    const int j = n0 + nt * 16 + nlane;
    const float bias = b_ih[j] + b_hh[j];
    #pragma unroll
    for (int v = 0; v < 8; ++v) acc[nt][v] = bias;
  }

  for (int k0 = 0; k0 < EE; k0 += 32) {
    v16bf a = load_a_tile(xb + (size_t)m0 * EE, EE, k0);
    #pragma unroll
    for (int nt = 0; nt < 4; ++nt) {
      v16bf b = load_b_tile(wih + (size_t)(n0 + nt * 16) * EE, EE, k0);
      acc[nt] = __builtin_amdgcn_wmma_f32_16x16x32_bf16(
          false, a, false, b, (short)0, acc[nt], false, false);
    }
  }

  const int mb = (l >> 4) * 8;  // C/D layout: lane group selects M half
  #pragma unroll
  for (int nt = 0; nt < 4; ++nt)
    #pragma unroll
    for (int v = 0; v < 8; ++v)
      xproj[(size_t)(m0 + mb + v) * HH + n0 + nt * 16 + nlane] = acc[nt][v];
}

// ---------------------------------------------------------------------------
// Kernel 3: persistent RNN scan. 64 blocks x 128 threads. Block owns columns
// [j0, j0+16). 4 waves split K=1024 into 256-chunks; LDS reduce; tanh; bf16
// ping-pong h; device-scope atomic grid barrier per step.
// ---------------------------------------------------------------------------
__global__ void __launch_bounds__(128)
rnn_scan(const unsigned short* __restrict__ whh,   // [H][H] bf16
         const float* __restrict__ xproj,          // [T*B][H] fp32
         unsigned short* __restrict__ h0buf,       // [B][H] bf16 (init 0)
         unsigned short* __restrict__ h1buf,       // [B][H] bf16
         int* __restrict__ counters) {             // [T] ints, init 0
  __shared__ float red[4 * 4 * 32 * 8];            // [wave][mtile][lane][8] = 16 KB
  const int w     = threadIdx.x >> 5;
  const int l     = threadIdx.x & 31;
  const int j0    = blockIdx.x * 16;
  const int nlane = l & 15;
  const int mb    = (l >> 4) * 8;
  const int kbase = w * 256;

  for (int t = 0; t < TT; ++t) {
    const unsigned short* hA = (t & 1) ? h1buf : h0buf;
    unsigned short*       hB = (t & 1) ? h0buf : h1buf;

    v8f acc[4];
    #pragma unroll
    for (int mt = 0; mt < 4; ++mt)
      #pragma unroll
      for (int v = 0; v < 8; ++v) acc[mt][v] = 0.0f;

    #pragma unroll
    for (int i = 0; i < 8; ++i) {
      const int k0 = kbase + i * 32;
      v16bf b = load_b_tile(whh + (size_t)j0 * HH, HH, k0);
      #pragma unroll
      for (int mt = 0; mt < 4; ++mt) {
        v16bf a = load_a_tile(hA + (size_t)(mt * 16) * HH, HH, k0);
        acc[mt] = __builtin_amdgcn_wmma_f32_16x16x32_bf16(
            false, a, false, b, (short)0, acc[mt], false, false);
      }
    }

    #pragma unroll
    for (int mt = 0; mt < 4; ++mt)
      #pragma unroll
      for (int v = 0; v < 8; ++v)
        red[(((w * 4) + mt) * 32 + l) * 8 + v] = acc[mt][v];
    __syncthreads();

    {   // wave w reduces m-tile mt == w across the 4 K-partials
      const int mt = w;
      #pragma unroll
      for (int v = 0; v < 8; ++v) {
        float s = 0.0f;
        #pragma unroll
        for (int wv = 0; wv < 4; ++wv)
          s += red[(((wv * 4) + mt) * 32 + l) * 8 + v];
        const int row = mt * 16 + mb + v;
        const float z = s + xproj[((size_t)t * BB + row) * HH + j0 + nlane];
        const float hv = tanhf(z);
        hB[(size_t)row * HH + j0 + nlane] = f32_to_bf16_raw(hv);
      }
    }

    // grid barrier: release stores, signal, spin, acquire
    __threadfence();
    __syncthreads();
    if (threadIdx.x == 0) {
      __hip_atomic_fetch_add(&counters[t], 1, __ATOMIC_RELEASE,
                             __HIP_MEMORY_SCOPE_AGENT);
      while (__hip_atomic_load(&counters[t], __ATOMIC_ACQUIRE,
                               __HIP_MEMORY_SCOPE_AGENT) < (int)gridDim.x) {
        __builtin_amdgcn_s_sleep(1);
      }
    }
    __syncthreads();
  }
}

// ---------------------------------------------------------------------------
// Kernel 4: y1 = relu(h_last @ fc1_w^T + fc1_b), fp32 out. 16 blocks x 128.
// h_last bf16 lives in h0buf after 512 (even count) steps.
// ---------------------------------------------------------------------------
__global__ void __launch_bounds__(128)
fc1_kernel(const unsigned short* __restrict__ hbf,   // [B][H] bf16
           const unsigned short* __restrict__ w1,    // [H][H] bf16
           const float* __restrict__ b1,
           float* __restrict__ y1f) {
  const int w  = threadIdx.x >> 5;
  const int l  = threadIdx.x & 31;
  const int m0 = w * 16;                  // M=64 -> one block row, wave per strip
  const int n0 = blockIdx.x * 64;
  const int nlane = l & 15;

  v8f acc[4];
  #pragma unroll
  for (int nt = 0; nt < 4; ++nt) {
    const float bias = b1[n0 + nt * 16 + nlane];
    #pragma unroll
    for (int v = 0; v < 8; ++v) acc[nt][v] = bias;
  }
  for (int k0 = 0; k0 < HH; k0 += 32) {
    v16bf a = load_a_tile(hbf + (size_t)m0 * HH, HH, k0);
    #pragma unroll
    for (int nt = 0; nt < 4; ++nt) {
      v16bf b = load_b_tile(w1 + (size_t)(n0 + nt * 16) * HH, HH, k0);
      acc[nt] = __builtin_amdgcn_wmma_f32_16x16x32_bf16(
          false, a, false, b, (short)0, acc[nt], false, false);
    }
  }
  const int mb = (l >> 4) * 8;
  #pragma unroll
  for (int nt = 0; nt < 4; ++nt)
    #pragma unroll
    for (int v = 0; v < 8; ++v) {
      const float y = fmaxf(acc[nt][v], 0.0f);
      y1f[(size_t)(m0 + mb + v) * HH + n0 + nt * 16 + nlane] = y;
    }
}

// ---------------------------------------------------------------------------
// Kernel 5: out[b][c] = y1[b] . fc2_w[c] + fc2_b[c]  (64x20, fp32 scalar path)
// ---------------------------------------------------------------------------
__global__ void fc2_kernel(const float* __restrict__ y1,
                           const float* __restrict__ w2,
                           const float* __restrict__ b2,
                           float* __restrict__ out) {
  const int c = blockIdx.x;    // 0..19
  const int b = threadIdx.x;   // 0..63
  const float* yr = y1 + (size_t)b * HH;
  const float* wr = w2 + (size_t)c * HH;
  float s = 0.0f;
  for (int k = 0; k < HH; k += 4) {
    float4 a = *(const float4*)(yr + k);
    float4 q = *(const float4*)(wr + k);
    s += a.x * q.x + a.y * q.y + a.z * q.z + a.w * q.w;
  }
  out[b * NCLS + c] = s + b2[c];
}

// ---------------------------------------------------------------------------
extern "C" void kernel_launch(void* const* d_in, const int* in_sizes, int n_in,
                              void* d_out, int out_size, void* d_ws, size_t ws_size,
                              hipStream_t stream) {
  const float* x_in  = (const float*)d_in[0];
  const float* W_ih  = (const float*)d_in[1];
  const float* W_hh  = (const float*)d_in[2];
  const float* b_ih  = (const float*)d_in[3];
  const float* b_hh  = (const float*)d_in[4];
  const float* fc1_w = (const float*)d_in[5];
  const float* fc1_b = (const float*)d_in[6];
  const float* fc2_w = (const float*)d_in[7];
  const float* fc2_b = (const float*)d_in[8];

  char* ws = (char*)d_ws;
  size_t off = 0;
  auto carve = [&](size_t bytes) -> void* {
    void* p = ws + off;
    off = (off + bytes + 255) & ~(size_t)255;
    return p;
  };
  int*            counters = (int*)           carve((size_t)TT * sizeof(int));
  unsigned short* whhb     = (unsigned short*)carve((size_t)HH * HH * 2);
  unsigned short* wihb     = (unsigned short*)carve((size_t)HH * EE * 2);
  unsigned short* fc1wb    = (unsigned short*)carve((size_t)HH * HH * 2);
  unsigned short* xb       = (unsigned short*)carve((size_t)TT * BB * EE * 2);
  float*          xproj    = (float*)         carve((size_t)TT * BB * HH * 4);
  unsigned short* h0buf    = (unsigned short*)carve((size_t)BB * HH * 2);
  unsigned short* h1buf    = (unsigned short*)carve((size_t)BB * HH * 2);
  float*          y1f      = (float*)         carve((size_t)BB * HH * 4);
  (void)ws_size; (void)in_sizes; (void)n_in; (void)out_size;  // ~174 MB used

  // 1) init + bf16 conversion + x transpose
  init_convert<<<2048, 256, 0, stream>>>(x_in, W_ih, W_hh, fc1_w,
                                         xb, wihb, whhb, fc1wb, h0buf, counters);
  // 2) batched input projection GEMM (M=32768, N=1024, K=512)
  dim3 gProj((TT * BB) / 64, HH / 64);
  xproj_gemm<<<gProj, 128, 0, stream>>>(xb, wihb, b_ih, b_hh, xproj);
  // 3) sequential recurrence, persistent grid with per-step barrier
  rnn_scan<<<HH / 16, 128, 0, stream>>>(whhb, xproj, h0buf, h1buf, counters);
  // 4) MLP head
  fc1_kernel<<<HH / 64, 128, 0, stream>>>(h0buf, fc1wb, fc1_b, y1f);
  fc2_kernel<<<NCLS, BB, 0, stream>>>(y1f, fc2_w, fc2_b, (float*)d_out);
}
